// Cheb_conv_53154515256266
// MI455X (gfx1250) — matrix operand order repaired and checked by reading, not compiled, not run on hardware
//
#include <hip/hip_runtime.h>

typedef __attribute__((ext_vector_type(16))) _Float16 v16h;
typedef __attribute__((ext_vector_type(8)))  float    v8f;
typedef __attribute__((ext_vector_type(4)))  unsigned v4u;
typedef __attribute__((ext_vector_type(8)))  int      v8i;
typedef __attribute__((ext_vector_type(4)))  int      v4i;

#define NV 24          // graph vertices
#define CIN 32         // input channels
#define COUT 32        // output channels
#define KORD 3         // Chebyshev order
#define BATCH 64
#define KDIM 768       // CIN*NV (GEMM K)
#define MDIM 768       // COUT*NV (GEMM M)
#define NDIM 512       // TLEN    (GEMM N)
#define LDST 40        // padded LDS row stride in halfs (32 + 8)
#define NKSTEP (KDIM / 32)          // 24 K-iterations
#define A_BUF_HALFS (128 * LDST)    // one A buffer: 5120 halfs = 10240 B
#define A_BUF_BYTES (A_BUF_HALFS * 2)

#define HAVE_TDM __has_builtin(__builtin_amdgcn_tensor_load_to_lds)

// ---------------------------------------------------------------------------
// Stage 0: normalized Laplacian + Chebyshev polynomials T[3][24][24]
// L[i][j] = adj[j][i] * dinvsqrt[i] * dinvsqrt[j];  T0=I, T1=L, T2=2*L@L - I
// ---------------------------------------------------------------------------
__global__ void cheb_T_kernel(const float* __restrict__ adj,
                              float* __restrict__ T) {
    __shared__ float sAdj[NV * NV];
    __shared__ float sDis[NV];
    __shared__ float sL[NV * NV];
    const int tid = threadIdx.x;
    if (tid < NV * NV) sAdj[tid] = adj[tid];
    __syncthreads();
    if (tid < NV) {
        float s = 0.f;
        for (int m = 0; m < NV; ++m) s += sAdj[tid * NV + m];
        sDis[tid] = (s > 0.f) ? rsqrtf(s) : 0.f;
    }
    __syncthreads();
    if (tid < NV * NV) {
        const int i = tid / NV, j = tid % NV;
        sL[tid] = sAdj[j * NV + i] * sDis[i] * sDis[j];
    }
    __syncthreads();
    if (tid < NV * NV) {
        const int i = tid / NV, j = tid % NV;
        float acc = 0.f;
        for (int p = 0; p < NV; ++p) acc += sL[i * NV + p] * sL[p * NV + j];
        const float eye = (i == j) ? 1.f : 0.f;
        T[0 * NV * NV + tid] = eye;
        T[1 * NV * NV + tid] = sL[tid];
        T[2 * NV * NV + tid] = 2.f * acc - eye;
    }
}

// ---------------------------------------------------------------------------
// Stage 1: fold graph operator: W2[(o,m)][(c,n)] = sum_k T[k][n][m]*Theta[k][c][o]
// Stored row-major 768x768 as f16 in workspace (L2-resident, 1.2 MB).
// ---------------------------------------------------------------------------
__global__ void w2_kernel(const float* __restrict__ T,
                          const float* __restrict__ Theta,
                          _Float16* __restrict__ W2) {
    const int idx = blockIdx.x * blockDim.x + threadIdx.x;   // < 768*768 exactly
    const int row = idx / KDIM;          // (o,m)
    const int col = idx - row * KDIM;    // (c,n)
    const int o = row / NV, m = row % NV;
    const int c = col / NV, n = col % NV;
    float v = 0.f;
#pragma unroll
    for (int k = 0; k < KORD; ++k)
        v += T[k * NV * NV + n * NV + m] * Theta[k * CIN * COUT + c * COUT + o];
    W2[idx] = (_Float16)v;
}

// ---------------------------------------------------------------------------
// Stage 2: batched GEMM  Out_b (768x512, f32) = W2 (768x768, f16) @ X_b (f32->f16)
// 256 threads (8 waves), 128x128 tile, wave = 32x64 region, K-step 32.
// A tile fetched by the Tensor Data Mover into double-buffered LDS with
// hardware padding to the 40-half stride; B converted f32->f16 via VALU.
// ---------------------------------------------------------------------------
__device__ inline unsigned pack2h(float a, float b) {
    union { _Float16 h[2]; unsigned u; } q;
    q.h[0] = (_Float16)a;
    q.h[1] = (_Float16)b;
    return q.u;
}

union FragU { uint4 q[2]; v16h v; };

#if HAVE_TDM
// Issue one TDM load of a 128-row x 32-half tile of W2 (row stride 768 halfs)
// into LDS at byte offset lds_off, padding each 16-DWORD row by 4 DWORDs
// (-> 40-half LDS stride). D# bitfields per CDNA5 ISA 8.3/8.4.
// amdgpu-toolchain builtin arity: (v4u, v8i, v4i, v4i, v8i, i32 cpol).
__device__ inline void tdm_load_A(const _Float16* gsrc, unsigned lds_off) {
    const unsigned long long ga = (unsigned long long)(uintptr_t)gsrc;
    v4u g0;
    g0[0] = 1u;                              // count=1 (valid), user mode
    g0[1] = lds_off;                         // lds_addr (bytes)
    g0[2] = (unsigned)ga;                    // global_addr[31:0]
    g0[3] = (unsigned)((ga >> 32) & 0x01FFFFFFu) | (2u << 30);  // addr[56:32] | type=2
    v8i g1;
    g1[0] = (1 << 16)        // data_size = 1 (2 bytes)
          | (1 << 20)        // pad_enable
          | (3 << 22)        // pad_interval: 16 DWORDs between pads
          | (3 << 25);       // pad_amount: 4 DWORDs
    g1[1] = (int)(768u << 16);   // tensor_dim0[15:0] = 768 (bits 63:48)
    g1[2] = (int)(768u << 16);   // tensor_dim0 hi=0; tensor_dim1[15:0]=768 (bits 95:80)
    g1[3] = (int)(32u << 16);    // tensor_dim1 hi=0; tile_dim0=32 (bits 127:112)
    g1[4] = 128;                 // tile_dim1=128 (bits 143:128); tile_dim2=0
    g1[5] = 768;                 // tensor_dim0_stride[31:0]
    g1[6] = 0;                   // stride hi; tensor_dim1_stride lo
    g1[7] = 0;
    v4i g2 = {0, 0, 0, 0};       // unused (2D tensor)
    v4i g3 = {0, 0, 0, 0};
    v8i g4 = {0, 0, 0, 0, 0, 0, 0, 0};   // extra group (unused here)
    __builtin_amdgcn_tensor_load_to_lds(g0, g1, g2, g3, g4, 0);
}
#endif

__global__ __launch_bounds__(256)
void cheb_gemm_kernel(const _Float16* __restrict__ W2,
                      const float* __restrict__ X,
                      float* __restrict__ Out) {
    // [ A buf 0 | A buf 1 | B ]  (A bufs at byte 0 / A_BUF_BYTES; B at 2*A_BUF_BYTES)
    __shared__ _Float16 smem[2 * A_BUF_HALFS + 128 * LDST];
    _Float16* const ldsB = smem + 2 * A_BUF_HALFS;

    const int tid    = threadIdx.x;
    const int b      = blockIdx.z;
    const int m_base = blockIdx.y * 128;
    const int t_base = blockIdx.x * 128;

    const float* __restrict__ Xb = X   + (size_t)b * KDIM * NDIM;
    float* __restrict__       Ob = Out + (size_t)b * MDIM * NDIM;
    const _Float16* __restrict__ Arow = W2 + (size_t)m_base * KDIM;

    const int lane = tid & 31;
    const int wid  = tid >> 5;
    const int rh   = lane & 15;     // row/col within 16
    const int hsel = lane >> 4;     // lane half select
    const int wm   = wid >> 1;      // 0..3 -> wave m offset wm*32
    const int wn   = wid & 1;       // 0..1 -> wave n offset wn*64

    v8f acc[2][4];
    {
        v8f z = {};
#pragma unroll
        for (int im = 0; im < 2; ++im)
#pragma unroll
            for (int jn = 0; jn < 4; ++jn) acc[im][jn] = z;
    }

    // B fill: thread owns one t column, 16 consecutive k values
    const int b_col  = tid & 127;   // t within tile
    const int b_kseg = tid >> 7;    // 0/1 -> k 0..15 / 16..31

#if !HAVE_TDM
    const int a_row = tid >> 1;     // fallback manual A staging
    const int a_seg = tid & 1;
#else
    if (wid == 0) tdm_load_A(Arow, 0u);          // prefetch tile 0 into buf 0
#endif

    for (int kt = 0; kt < NKSTEP; ++kt) {
        const int k0 = kt * 32;

        // ---- global loads of x into registers ----
        float xr[16];
#pragma unroll
        for (int i = 0; i < 16; ++i)
            xr[i] = Xb[(size_t)(k0 + b_kseg * 16 + i) * NDIM + t_base + b_col];

#if !HAVE_TDM
        const uint4* pa = reinterpret_cast<const uint4*>(
            Arow + (size_t)a_row * KDIM + k0 + a_seg * 16);
        const uint4 ga0 = pa[0];
        const uint4 ga1 = pa[1];
#endif

        __syncthreads();   // previous iteration's LDS reads done

#if HAVE_TDM
        if (wid == 0 && kt + 1 < NKSTEP)         // A tile kt+1 -> other buffer
            tdm_load_A(Arow + (kt + 1) * 32, (unsigned)(((kt + 1) & 1) * A_BUF_BYTES));
#else
        _Float16* ldsA0 = smem;                  // single-buffer fallback
        *reinterpret_cast<uint4*>(&ldsA0[a_row * LDST + a_seg * 16])     = ga0;
        *reinterpret_cast<uint4*>(&ldsA0[a_row * LDST + a_seg * 16 + 8]) = ga1;
#endif

        // ---- convert + store B tile transposed: ldsB[t][k] ----
        uint4 u0, u1;
        u0.x = pack2h(xr[0],  xr[1]);  u0.y = pack2h(xr[2],  xr[3]);
        u0.z = pack2h(xr[4],  xr[5]);  u0.w = pack2h(xr[6],  xr[7]);
        u1.x = pack2h(xr[8],  xr[9]);  u1.y = pack2h(xr[10], xr[11]);
        u1.z = pack2h(xr[12], xr[13]); u1.w = pack2h(xr[14], xr[15]);
        *reinterpret_cast<uint4*>(&ldsB[b_col * LDST + b_kseg * 16])     = u0;
        *reinterpret_cast<uint4*>(&ldsB[b_col * LDST + b_kseg * 16 + 8]) = u1;

#if HAVE_TDM
        if (wid == 0) {                           // current tile done; next stays in flight
            if (kt + 1 < NKSTEP) __builtin_amdgcn_s_wait_tensorcnt((short)1);
            else                 __builtin_amdgcn_s_wait_tensorcnt((short)0);
        }
        const _Float16* ldsA = smem + (kt & 1) * A_BUF_HALFS;
#else
        const _Float16* ldsA = smem;
#endif

        __syncthreads();   // tiles visible to all waves

        // ---- fragments ----
        // A (16x32 f16): lanes 0-15 hold K {0..7,16..23}, lanes 16-31 K {8..15,24..31}
        v16h afrag[2];
#pragma unroll
        for (int im = 0; im < 2; ++im) {
            const uint4* p = reinterpret_cast<const uint4*>(
                &ldsA[(wm * 32 + im * 16 + rh) * LDST]);
            FragU u;
            u.q[0] = p[hsel];
            u.q[1] = p[2 + hsel];
            afrag[im] = u.v;
        }
        // B (32x16 f16): lane holds column rh; lanes 0-15 K {0..7,16..23}, lanes 16-31 K {8..15,24..31}
        v16h bfrag[4];
#pragma unroll
        for (int jn = 0; jn < 4; ++jn) {
            const uint4* p = reinterpret_cast<const uint4*>(
                &ldsB[(wn * 64 + jn * 16 + rh) * LDST]);
            FragU u;
            u.q[0] = p[hsel];
            u.q[1] = p[2 + hsel];
            bfrag[jn] = u.v;
        }

        // ---- 8 WMMAs: 32x64 region per wave ----
#pragma unroll
        for (int im = 0; im < 2; ++im)
#pragma unroll
            for (int jn = 0; jn < 4; ++jn)
                acc[im][jn] = __builtin_amdgcn_wmma_f32_16x16x32_f16(
                    false, afrag[im], false, bfrag[jn],
                    (short)0, acc[im][jn], false, false);
    }

    // ---- epilogue: C/D layout -> element (M=j+8*hsel, N=rh) in VGPR j ----
#pragma unroll
    for (int im = 0; im < 2; ++im) {
#pragma unroll
        for (int jn = 0; jn < 4; ++jn) {
            const int M0 = m_base + wm * 32 + im * 16;
            const int N0 = t_base + wn * 64 + jn * 16;
#pragma unroll
            for (int j = 0; j < 8; ++j)
                Ob[(size_t)(M0 + j + 8 * hsel) * NDIM + N0 + rh] = acc[im][jn][j];
        }
    }
}

// ---------------------------------------------------------------------------
// Launch
// ---------------------------------------------------------------------------
extern "C" void kernel_launch(void* const* d_in, const int* in_sizes, int n_in,
                              void* d_out, int out_size, void* d_ws, size_t ws_size,
                              hipStream_t stream) {
    const float* x     = (const float*)d_in[0];   // (64,32,24,512)
    const float* adj   = (const float*)d_in[1];   // (24,24)
    const float* Theta = (const float*)d_in[2];   // (3,32,32)
    float* out = (float*)d_out;                   // (64,32,24,512)

    _Float16* W2 = (_Float16*)d_ws;                                   // 768*768 f16
    float*    T  = (float*)((char*)d_ws + (size_t)MDIM * KDIM * 2);   // 3*24*24 f32

    cheb_T_kernel<<<1, 576, 0, stream>>>(adj, T);
    w2_kernel<<<(MDIM * KDIM) / 256, 256, 0, stream>>>(T, Theta, W2);

    dim3 grid(NDIM / 128, MDIM / 128, BATCH);     // (4, 6, 64)
    cheb_gemm_kernel<<<grid, 256, 0, stream>>>(W2, x, out);
}
